// ABC_2D_Large_86904368267933
// MI455X (gfx1250) — compile-verified, matrix operand order checked by reading, not run on hardware
//
#include <hip/hip_runtime.h>
#include <hip/hip_bf16.h>

typedef __attribute__((ext_vector_type(16))) _Float16 v16h;
typedef __attribute__((ext_vector_type(2)))  _Float16 v2h;
typedef __attribute__((ext_vector_type(8)))  float    v8f;
typedef __attribute__((ext_vector_type(4)))  float    v4f;

#define BB    16
#define CC    64
#define NHW   4096    // NH*NW
#define KL    25
#define KK    1600    // C*KL
#define MM    256     // OUT
#define KT    32      // K per WMMA step (f16)
#define NT    64      // N columns per workgroup
#define MW    32      // M rows per wave
#define NSTEPS (KK / KT)            // 50
#define WF_HALVES (NSTEPS * 16 * 32 * 16)   // 409600 halves = 800 KB

// ---------------------------------------------------------------------------
// One-time pre-pass: weights f32 -> f16, laid out in WMMA A-fragment order:
// wf[((s*16 + mtile)*32 + lane)*16 + slot], slot j -> k = s*32 + (j<8?j:j+8) + 8*(lane>>4)
// Main loop A-load becomes one contiguous 32B read per m-tile per step.
__launch_bounds__(256)
__global__ void convert_weights_kernel(const float* __restrict__ w,
                                       _Float16* __restrict__ wf)
{
    const int gid   = blockIdx.x * 256 + threadIdx.x;  // 204800 threads, 1 slot-pair each
    const int p     = gid & 7;                          // slot pair (j = 2p, 2p+1)
    const int lane  = (gid >> 3) & 31;
    const int mtile = (gid >> 8) & 15;
    const int s     = gid >> 12;                        // 0..49
    const int lm    = lane & 15;
    const int lhi   = lane >> 4;
    const int m     = mtile * 16 + lm;
    const int j     = 2 * p;
    const int k     = s * 32 + (j < 8 ? j : j + 8) + 8 * lhi;

    const float* src = w + m * KK + k;                  // 8B aligned (k even)
    v2h h;
    h[0] = (_Float16)src[0];
    h[1] = (_Float16)src[1];
    *(v2h*)(wf + gid * 2) = h;
}

// ---------------------------------------------------------------------------
// B-fragment LDS staging (double buffered, 8 KB):
// lane = (n%16) + 16*((k>>3)&1), slot = (k&7) + 8*((k>>4)&1)
__shared__ __align__(32) _Float16 Bs[2][4][32][16];

__device__ __forceinline__ void stage_tile(int k0, int buf,
                                           int gk, int gn_base,
                                           int dst_hi, int slot,
                                           int b, int nh,
                                           const int* __restrict__ conv_hash,
                                           const unsigned char* __restrict__ zerofy,
                                           const float* __restrict__ x)
{
    const int k_abs = k0 + gk;
    const int c     = k_abs / KL;
    const int kl    = k_abs - c * KL;
    int hidx = (((b * CC + c) * 64 + nh) * 64 + gn_base) * KL + kl;

    int           idx[8];
    unsigned char z[8];
#pragma unroll
    for (int it = 0; it < 8; ++it) {                    // streamed once: NT hint
        idx[it] = __builtin_nontemporal_load(&conv_hash[hidx]);
        z[it]   = __builtin_nontemporal_load(&zerofy[hidx]);
        hidx   += 8 * KL;
    }

    float xv[8];
#pragma unroll
    for (int it = 0; it < 8; ++it)                      // unconditional, L0/L2-hot
        xv[it] = x[idx[it]];

    _Float16* dst = &Bs[buf][0][dst_hi + gn_base][slot];
#pragma unroll
    for (int it = 0; it < 8; ++it) {                    // cndmask, no branch
        const float v = z[it] ? 0.0f : xv[it];
        dst[512 * (it >> 1) + 128 * (it & 1)] = (_Float16)v;
    }
}

template <bool PRECONV>
__launch_bounds__(256)
__global__ void im2col_wmma_f16_kernel(const float* __restrict__ x,
                                       const int* __restrict__ conv_hash,
                                       const unsigned char* __restrict__ zerofy,
                                       const float* __restrict__ weights,
                                       const _Float16* __restrict__ wf,
                                       float* __restrict__ out)
{
    const int tid  = threadIdx.x;
    const int lane = tid & 31;
    const int wave = tid >> 5;
    const int lm   = lane & 15;
    const int lhi  = lane >> 4;

    const int n0 = blockIdx.x * NT;
    const int b  = n0 >> 12;
    const int nh = (n0 >> 6) & 63;

    const int gk      = tid & 31;
    const int gn_base = tid >> 5;
    const int dst_hi  = 16 * ((gk >> 3) & 1);
    const int slot    = (gk & 7) + 8 * ((gk >> 4) & 1);

    const int m0 = wave * MW;

    v8f acc[2][4];
#pragma unroll
    for (int i = 0; i < 2; ++i)
#pragma unroll
        for (int j = 0; j < 4; ++j)
            acc[i][j] = v8f{0.f,0.f,0.f,0.f,0.f,0.f,0.f,0.f};

    stage_tile(0, 0, gk, gn_base, dst_hi, slot, b, nh, conv_hash, zerofy, x);
    __syncthreads();

    for (int s = 0; s < NSTEPS; ++s) {
        const int k0  = s * KT;
        const int cur = s & 1;

        if (s + 1 < NSTEPS)
            stage_tile(k0 + KT, cur ^ 1, gk, gn_base, dst_hi, slot,
                       b, nh, conv_hash, zerofy, x);

        v16h afrag[2];
        if (PRECONV) {
            // fragment-ordered f16 table in workspace: one 32B load per m-tile
#pragma unroll
            for (int mt = 0; mt < 2; ++mt)
                afrag[mt] = *(const v16h*)(wf +
                    (((s * 16) + (m0 >> 4) + mt) * 32 + lane) * 16);
        } else {
            // fallback: inline f32 -> f16 conversion from raw weights
#pragma unroll
            for (int mt = 0; mt < 2; ++mt) {
                const int m = m0 + mt * 16 + lm;
                const float* wrow = weights + m * KK + k0 + 8 * lhi;
                const v4f w0 = *(const v4f*)(wrow);
                const v4f w1 = *(const v4f*)(wrow + 4);
                const v4f w2 = *(const v4f*)(wrow + 16);
                const v4f w3 = *(const v4f*)(wrow + 20);
                v16h a;
#pragma unroll
                for (int j = 0; j < 4; ++j) {
                    a[j]      = (_Float16)w0[j];
                    a[4 + j]  = (_Float16)w1[j];
                    a[8 + j]  = (_Float16)w2[j];
                    a[12 + j] = (_Float16)w3[j];
                }
                afrag[mt] = a;
            }
        }

#pragma unroll
        for (int nt = 0; nt < 4; ++nt) {
            const v16h bfrag = *(const v16h*)(&Bs[cur][nt][lane][0]);
#pragma unroll
            for (int mt = 0; mt < 2; ++mt) {
                acc[mt][nt] = __builtin_amdgcn_wmma_f32_16x16x32_f16(
                    false, afrag[mt], false, bfrag,
                    (short)0, acc[mt][nt], false, false);
            }
        }
        __syncthreads();
    }

    // store D: out[b][m][nh][nw]; VGPR r -> m = r + 8*lhi, n = lm
#pragma unroll
    for (int mt = 0; mt < 2; ++mt) {
#pragma unroll
        for (int nt = 0; nt < 4; ++nt) {
            const int nw    = nt * 16 + lm;
            const int obase = b * (MM * NHW) + nh * 64 + nw;
#pragma unroll
            for (int r = 0; r < 8; ++r) {
                const int m = m0 + mt * 16 + 8 * lhi + r;
                __builtin_nontemporal_store(acc[mt][nt][r], &out[obase + m * NHW]);
            }
        }
    }
}

extern "C" void kernel_launch(void* const* d_in, const int* in_sizes, int n_in,
                              void* d_out, int out_size, void* d_ws, size_t ws_size,
                              hipStream_t stream) {
    const float*         x         = (const float*)d_in[0];
    const int*           conv_hash = (const int*)d_in[1];
    const unsigned char* zerofy    = (const unsigned char*)d_in[2]; // bool, 1B/elem
    const float*         weights   = (const float*)d_in[3];
    float*               out       = (float*)d_out;

    const int n_cols = BB * NHW;                 // 65536
    dim3 grid(n_cols / NT);                      // 1024 workgroups
    dim3 block(256);                             // 8 waves, wave32

    if (ws_size >= (size_t)WF_HALVES * sizeof(_Float16)) {
        _Float16* wf = (_Float16*)d_ws;
        convert_weights_kernel<<<dim3(WF_HALVES / 2 / 256), block, 0, stream>>>(weights, wf);
        im2col_wmma_f16_kernel<true><<<grid, block, 0, stream>>>(
            x, conv_hash, zerofy, weights, wf, out);
    } else {
        im2col_wmma_f16_kernel<false><<<grid, block, 0, stream>>>(
            x, conv_hash, zerofy, weights, (const _Float16*)nullptr, out);
    }
}
